// TemporalGraphNetwork_17540646437557
// MI455X (gfx1250) — matrix operand order repaired and checked by reading, not compiled, not run on hardware
//
#include <hip/hip_runtime.h>
#include <math.h>
#include <stdint.h>

// ---------------- problem constants (match reference) ----------------
static constexpr int N_   = 50000;   // nodes
static constexpr int E_   = 40000;   // edges
static constexpr int BATCH = 200;
static constexpr int NB    = E_ / BATCH;   // 200 sequential steps
static constexpr int ND = 256, ED = 128, TD = 128, MD = 256, DD = 256;
static constexpr int P_  = 4096;
static constexpr int SORT_N = 65536;       // next pow2 >= E_
static constexpr int BR = 208;             // padded batch rows = 13*16

// ---------------- WMMA helpers (v_wmma_f32_16x16x32_f16) ----------------
typedef __attribute__((ext_vector_type(16))) _Float16 v16h;
typedef __attribute__((ext_vector_type(8)))  float    v8f;

__device__ inline v8f wmma_f16(v16h a, v16h b, v8f c) {
  return __builtin_amdgcn_wmma_f32_16x16x32_f16(false, a, false, b, (short)0, c,
                                                false, false);
}

// A fragment: 16x32 f16.  lane m(0..15) holds row m, K = {kb..kb+7, 16+kb..16+kb+7}
template <typename GA>
__device__ inline v16h load_afrag(GA ga, int lane) {
  v16h a;
  const int m  = lane & 15;
  const int kb = (lane >> 4) * 8;
#pragma unroll
  for (int i = 0; i < 8; ++i) {
    a[i]     = (_Float16)ga(m, kb + i);
    a[i + 8] = (_Float16)ga(m, 16 + kb + i);
  }
  return a;
}

// B fragment: 32x16 f16.  lane n(0..15) holds col n, K = {kb..kb+7, 16+kb..16+kb+7}
template <typename GB>
__device__ inline v16h load_bfrag(GB gb, int lane) {
  v16h b;
  const int n  = lane & 15;
  const int kb = (lane >> 4) * 8;
#pragma unroll
  for (int i = 0; i < 8; ++i) {
    b[i]     = (_Float16)gb(kb + i, n);
    b[i + 8] = (_Float16)gb(16 + kb + i, n);
  }
  return b;
}

// C/D: VGPR v holds row v + 8*(lane>>4), col lane&15
template <typename PC>
__device__ inline void store_cfrag(PC pc, int lane, v8f c) {
  const int n  = lane & 15;
  const int mb = (lane >> 4) * 8;
#pragma unroll
  for (int v = 0; v < 8; ++v) pc(mb + v, n, c[v]);
}

// One wave computes a 16 x (GN*16) row-group: A fragment reused GN times.
template <int GN, typename GA, typename GB, typename PC>
__device__ inline void wmma_rowgroup(int K, GA ga, GB gb, PC pc, int lane) {
  v8f acc[GN];
#pragma unroll
  for (int g = 0; g < GN; ++g) acc[g] = (v8f){};
  for (int k0 = 0; k0 < K; k0 += 32) {
    v16h a = load_afrag([&](int m, int kl) { return ga(m, k0 + kl); }, lane);
#pragma unroll
    for (int g = 0; g < GN; ++g) {
      v16h b = load_bfrag(
          [&](int kl, int n) { return gb(k0 + kl, g * 16 + n); }, lane);
      acc[g] = wmma_f16(a, b, acc[g]);
    }
  }
#pragma unroll
  for (int g = 0; g < GN; ++g)
    store_cfrag([&](int m, int n, float v) { pc(m, g * 16 + n, v); }, lane,
                acc[g]);
}

// order-preserving float <-> u32 (for atomicMax-based segment max / sort keys)
__device__ inline unsigned fenc(float f) {
  unsigned u = __float_as_uint(f);
  return (u & 0x80000000u) ? ~u : (u | 0x80000000u);
}
__device__ inline float fdec(unsigned u) {
  unsigned b = (u & 0x80000000u) ? (u & 0x7fffffffu) : ~u;
  return __uint_as_float(b);
}
static constexpr unsigned ENC_NEG_INF = 0x007FFFFFu; // fenc(-inf)

// ---------------- small utility kernels ----------------
__global__ void k_fill_f32(float* p, float v, long long n) {
  long long i = (long long)blockIdx.x * blockDim.x + threadIdx.x;
  if (i < n) p[i] = v;
}
__global__ void k_fill_f16(_Float16* p, float v, long long n) {
  long long i = (long long)blockIdx.x * blockDim.x + threadIdx.x;
  if (i < n) p[i] = (_Float16)v;
}
__global__ void k_fill_u32(unsigned* p, unsigned v, long long n) {
  long long i = (long long)blockIdx.x * blockDim.x + threadIdx.x;
  if (i < n) p[i] = v;
}
__global__ void k_cvt_f16(const float* s, _Float16* d, long long n) {
  long long i = (long long)blockIdx.x * blockDim.x + threadIdx.x;
  if (i < n) d[i] = (_Float16)s[i];
}

__global__ void k_init_keys(const float* t, unsigned long long* keys) {
  int i = blockIdx.x * blockDim.x + threadIdx.x;
  if (i >= SORT_N) return;
  keys[i] = (i < E_) ? (((unsigned long long)fenc(t[i]) << 32) | (unsigned)i)
                     : ~0ull;
}
__global__ void k_bitonic(unsigned long long* keys, int j, int k) {
  int i = blockIdx.x * blockDim.x + threadIdx.x;
  int l = i ^ j;
  if (l > i && i < SORT_N) {
    unsigned long long a = keys[i], b = keys[l];
    bool up = ((i & k) == 0);
    if ((a > b) == up) { keys[i] = b; keys[l] = a; }
  }
}
__global__ void k_extract_order(const unsigned long long* keys, int* order) {
  int i = blockIdx.x * blockDim.x + threadIdx.x;
  if (i < E_) order[i] = (int)(keys[i] & 0xffffffffu);
}

__global__ void k_time_enc(const float* t, const float* w, const float* bb,
                           float* te) {
  long long idx = (long long)blockIdx.x * blockDim.x + threadIdx.x;
  if (idx >= (long long)E_ * TD) return;
  int e = (int)(idx / TD), j = (int)(idx % TD);
  te[idx] = sinf(t[e] * w[j] + bb[j]);
}
__global__ void k_tmax(const float* t, unsigned* tmaxb) {
  int e = blockIdx.x * blockDim.x + threadIdx.x;
  if (e < E_) atomicMax(tmaxb, fenc(t[e]));
}
__global__ void k_gte(const unsigned* tmaxb, const float* w, const float* bb,
                      float* gte) {
  int j = threadIdx.x;
  if (j < TD) gte[j] = sinf(fdec(*tmaxb) * w[j] + bb[j]);
}

// last-write-wins flags within each batch of 200 sorted edges
__global__ void k_writer(const int* order, const int* ei0, int* writer) {
  int i = blockIdx.x * blockDim.x + threadIdx.x;
  if (i >= E_) return;
  int be = (i / BATCH) * BATCH + BATCH;
  int my = ei0[order[i]];
  int w = 1;
  for (int j = i + 1; j < be; ++j)
    if (ei0[order[j]] == my) { w = 0; break; }
  writer[i] = w;
}

// ---------------- big parallel WMMA GEMMs (16x64 per wave) ----------------
// pre1[i,:] = bm1 + nf[src]@Wm1[0:256] + ea@Wm1[512:640] + te@Wm1[640:768]
__global__ void k_pre1(const float* nf, const int* ei0, const float* ea,
                       const float* te, const int* order,
                       const _Float16* Wm1h, const float* bm1, float* pre1) {
  int gw = (blockIdx.x * blockDim.x + threadIdx.x) >> 5;
  int lane = threadIdx.x & 31;
  const int Mt = E_ / 16;
  if (gw >= Mt * 4) return;
  int mt = gw >> 2, nb = (gw & 3) * 64;
  wmma_rowgroup<4>(
      512,
      [&](int m, int kk) -> float {
        int i = mt * 16 + m;
        int eo = order[i];
        if (kk < 256) return nf[(long long)ei0[eo] * ND + kk];
        if (kk < 384) return ea[(long long)eo * ED + (kk - 256)];
        return te[(long long)eo * TD + (kk - 384)];
      },
      [&](int kk, int n) -> _Float16 {
        int r = (kk < 256) ? kk : kk + 256;   // skip Wm1 h_prev rows
        return Wm1h[(long long)r * DD + nb + n];
      },
      [&](int m, int n, float v) {
        pre1[(long long)(mt * 16 + m) * DD + nb + n] = v + bm1[nb + n];
      },
      lane);
}

// h = concat(nf, mem, gte) @ Wg        [N x 640] x [640 x 256]
__global__ void k_hfeat(const float* nf, const _Float16* memh,
                        const float* gte, const _Float16* Wgh, float* hfeat) {
  int gw = (blockIdx.x * blockDim.x + threadIdx.x) >> 5;
  int lane = threadIdx.x & 31;
  const int Mt = N_ / 16;
  if (gw >= Mt * 4) return;
  int mt = gw >> 2, nb = (gw & 3) * 64;
  __builtin_prefetch(Wgh + nb, 0, 3);
  wmma_rowgroup<4>(
      640,
      [&](int m, int kk) -> _Float16 {
        int n = mt * 16 + m;
        if (kk < 256) return (_Float16)nf[(long long)n * ND + kk];
        if (kk < 512) return memh[(long long)n * MD + (kk - 256)];
        return (_Float16)gte[kk - 512];
      },
      [&](int kk, int n) -> _Float16 {
        return Wgh[(long long)kk * DD + nb + n];
      },
      [&](int m, int n, float v) {
        hfeat[(long long)(mt * 16 + m) * DD + nb + n] = v;
      },
      lane);
}

// ef = concat(edge_attr, te) @ We       [E x 256] x [256 x 256]
__global__ void k_ef(const float* ea, const float* te, const _Float16* Weh,
                     float* ef) {
  int gw = (blockIdx.x * blockDim.x + threadIdx.x) >> 5;
  int lane = threadIdx.x & 31;
  const int Mt = E_ / 16;
  if (gw >= Mt * 4) return;
  int mt = gw >> 2, nb = (gw & 3) * 64;
  wmma_rowgroup<4>(
      256,
      [&](int m, int kk) -> float {
        int e = mt * 16 + m;
        if (kk < 128) return ea[(long long)e * ED + kk];
        return te[(long long)e * TD + (kk - 128)];
      },
      [&](int kk, int n) -> _Float16 {
        return Weh[(long long)kk * DD + nb + n];
      },
      [&](int m, int n, float v) {
        ef[(long long)(mt * 16 + m) * DD + nb + n] = v;
      },
      lane);
}

// -------- sequential TGN memory scan: one WGP, 32 waves ---------------
// h_prev gathered with GLOBAL_LOAD_ASYNC_TO_LDS_B128 (ASYNCcnt), f16 memory.
__global__ void __launch_bounds__(1024) k_scan(
    const int* order, const int* ei0, const int* writer, _Float16* memh,
    const float* pre1, const _Float16* Wm1h, const _Float16* Wm2h,
    const float* bm2, const _Float16* Wih, const _Float16* Whh,
    const float* bi, const float* bh, _Float16* uact, _Float16* msgb,
    float* gi, float* gh) {
  __shared__ __align__(16) _Float16 hprev[BR * MD];  // 104 KB LDS
  const int tid = threadIdx.x;
  const int nthr = 1024;
  const int wid = tid >> 5, lane = tid & 31, nw = 32;

  for (int b = 0; b < NB; ++b) {
    // --- async gather h_prev = mem[src] straight into LDS ---
    // one wave per row: 32 lanes x 16B = 512B = one f16 row of 256.
    // pad rows 200..207 stay garbage: they only feed discarded output rows.
    for (int r = wid; r < BATCH; r += nw) {
      int eo = order[b * BATCH + r];
      unsigned long long ga =
          (unsigned long long)(uintptr_t)(memh + (long long)ei0[eo] * MD) +
          (unsigned)(lane * 16);
      unsigned la =
          (unsigned)(uintptr_t)(hprev + r * MD) + (unsigned)(lane * 16);
      asm volatile("global_load_async_to_lds_b128 %0, %1, off"
                   :: "v"(la), "v"(ga) : "memory");
    }
    asm volatile("s_wait_asynccnt 0x0" ::: "memory");
    __syncthreads();

    // --- u = relu(pre1 + h_prev @ Wm1[256:512]) : 13 x 4 groups ---
    for (int t = wid; t < 13 * 4; t += nw) {
      int mt = t >> 2, nb = (t & 3) * 64;
      wmma_rowgroup<4>(
          MD,
          [&](int m, int kk) -> _Float16 {
            return hprev[(mt * 16 + m) * MD + kk];
          },
          [&](int kk, int n) -> _Float16 {
            return Wm1h[(long long)(256 + kk) * DD + nb + n];
          },
          [&](int m, int n, float v) {
            int r = mt * 16 + m;
            float u = 0.f;
            if (r < BATCH)
              u = fmaxf(v + pre1[(long long)(b * BATCH + r) * DD + nb + n], 0.f);
            uact[r * DD + nb + n] = (_Float16)u;
          },
          lane);
    }
    __threadfence(); __syncthreads();

    // --- msg = relu(u @ Wm2 + bm2) : 13 x 4 groups ---
    for (int t = wid; t < 13 * 4; t += nw) {
      int mt = t >> 2, nb = (t & 3) * 64;
      wmma_rowgroup<4>(
          DD,
          [&](int m, int kk) -> _Float16 {
            return uact[(mt * 16 + m) * DD + kk];
          },
          [&](int kk, int n) -> _Float16 {
            return Wm2h[(long long)kk * MD + nb + n];
          },
          [&](int m, int n, float v) {
            int r = mt * 16 + m;
            float u = 0.f;
            if (r < BATCH) u = fmaxf(v + bm2[nb + n], 0.f);
            msgb[r * MD + nb + n] = (_Float16)u;
          },
          lane);
    }
    __threadfence(); __syncthreads();

    // --- gi = msg @ Wi + bi ; gh = h_prev @ Wh + bh : 2 * 13 x 12 groups ---
    for (int t = wid; t < 2 * 13 * 12; t += nw) {
      bool isGi = (t < 13 * 12);
      int tt = isGi ? t : t - 13 * 12;
      int mt = tt / 12, nb = (tt % 12) * 64;
      const _Float16* W = isGi ? Wih : Whh;
      const float* bias = isGi ? bi : bh;
      float* outp = isGi ? gi : gh;
      wmma_rowgroup<4>(
          MD,
          [&](int m, int kk) -> _Float16 {
            return isGi ? msgb[(mt * 16 + m) * MD + kk]
                        : hprev[(mt * 16 + m) * MD + kk];
          },
          [&](int kk, int n) -> _Float16 {
            return W[(long long)kk * (3 * MD) + nb + n];
          },
          [&](int m, int n, float v) {
            outp[(mt * 16 + m) * (3 * MD) + nb + n] = v + bias[nb + n];
          },
          lane);
    }
    __threadfence(); __syncthreads();

    // --- GRU elementwise + last-write-wins scatter into mem (f16) ---
    for (int idx = tid; idx < BATCH * MD; idx += nthr) {
      int r = idx / MD, c = idx % MD;
      float ir = gi[r * 768 + c], iz = gi[r * 768 + 256 + c],
            in = gi[r * 768 + 512 + c];
      float hr = gh[r * 768 + c], hz = gh[r * 768 + 256 + c],
            hn = gh[r * 768 + 512 + c];
      float rg = 1.f / (1.f + expf(-(ir + hr)));
      float z  = 1.f / (1.f + expf(-(iz + hz)));
      float n  = tanhf(in + rg * hn);
      float hp = (float)hprev[r * MD + c];
      float hnew = (1.f - z) * n + z * hp;
      int i = b * BATCH + r;
      if (writer[i]) memh[(long long)ei0[order[i]] * MD + c] = (_Float16)hnew;
    }
    __threadfence(); __syncthreads();
  }
}

// ---------------- attention softmax / aggregation ----------------
__global__ void k_node_dots(const float* hfeat, const float* a_src,
                            const float* a_dst, float* asrcn, float* adstn) {
  int n = blockIdx.x * blockDim.x + threadIdx.x;
  if (n >= N_) return;
  float s1 = 0.f, s2 = 0.f;
  const float* hr = hfeat + (long long)n * DD;
  for (int j = 0; j < DD; ++j) { float h = hr[j]; s1 += h * a_src[j]; s2 += h * a_dst[j]; }
  asrcn[n] = s1; adstn[n] = s2;
}

__global__ void k_logit(const int* ei0, const int* ei1, const float* ef,
                        const float* a_e, const float* asrcn,
                        const float* adstn, float* logit, unsigned* mxenc) {
  int e = blockIdx.x * blockDim.x + threadIdx.x;
  if (e >= E_) return;
  int s = ei0[e], d = ei1[e];
  float l = asrcn[s] + adstn[d];
  const float* er = ef + (long long)e * DD;
  for (int j = 0; j < DD; ++j) l += er[j] * a_e[j];
  l = (l > 0.f) ? l : 0.2f * l;   // leaky_relu(0.2)
  logit[e] = l;
  atomicMax(&mxenc[d], fenc(l));
}

__global__ void k_alpha(const int* ei1, const float* logit,
                        const unsigned* mxenc, float* alpha, float* denom) {
  int e = blockIdx.x * blockDim.x + threadIdx.x;
  if (e >= E_) return;
  int d = ei1[e];
  float mx = fdec(mxenc[d]);
  if (!(mx > -1e30f)) mx = 0.f;     // isfinite -> else 0
  float a = expf(logit[e] - mx);
  alpha[e] = a;
  unsafeAtomicAdd(&denom[d], a);
}

__global__ void k_scatter(const int* ei0, const int* ei1, const float* alpha,
                          const float* hfeat, const float* ef, float* outacc) {
  long long idx = (long long)blockIdx.x * blockDim.x + threadIdx.x;
  if (idx >= (long long)E_ * 64) return;
  int e = (int)(idx >> 6);
  int c0 = ((int)(idx & 63)) << 2;
  int s = ei0[e], d = ei1[e];
  float a = alpha[e];
  const float* hr = hfeat + (long long)s * DD + c0;
  const float* er = ef + (long long)e * DD + c0;
  float* orow = outacc + (long long)d * DD + c0;
#pragma unroll
  for (int c = 0; c < 4; ++c) unsafeAtomicAdd(&orow[c], a * (hr[c] + er[c]));
}

__global__ void k_norm(float* outacc, const float* denom) {
  long long idx = (long long)blockIdx.x * blockDim.x + threadIdx.x;
  if (idx >= (long long)N_ * DD) return;
  outacc[idx] = outacc[idx] / (denom[idx / DD] + 1e-16f);
}

// ---------------- classifier head ----------------
__global__ void k_cls1(const float* outacc, const int* post,
                       const _Float16* Wc1h, const float* bc1, float* q) {
  int gw = (blockIdx.x * blockDim.x + threadIdx.x) >> 5;
  int lane = threadIdx.x & 31;
  const int Mt = P_ / 16;
  if (gw >= Mt * 2) return;
  int mt = gw >> 1, nb = (gw & 1) * 64;
  wmma_rowgroup<4>(
      DD,
      [&](int m, int kk) -> float {
        return outacc[(long long)post[mt * 16 + m] * DD + kk];
      },
      [&](int kk, int n) -> _Float16 {
        return Wc1h[(long long)kk * 128 + nb + n];
      },
      [&](int m, int n, float v) {
        q[(long long)(mt * 16 + m) * 128 + nb + n] = fmaxf(v + bc1[nb + n], 0.f);
      },
      lane);
}

__global__ void k_cls2(const float* q, const float* Wc2, const float* bc2,
                       float* out) {
  int p = blockIdx.x * blockDim.x + threadIdx.x;
  if (p >= P_) return;
  float s = bc2[0];
  const float* qr = q + (long long)p * 128;
  for (int j = 0; j < 128; ++j) s += qr[j] * Wc2[j];
  out[p] = 1.f / (1.f + expf(-s));
}

// ---------------- host orchestration ----------------
extern "C" void kernel_launch(void* const* d_in, const int* in_sizes, int n_in,
                              void* d_out, int out_size, void* d_ws,
                              size_t ws_size, hipStream_t stream) {
  const float* nf     = (const float*)d_in[0];
  const int*   ei     = (const int*)d_in[1];
  const float* ea     = (const float*)d_in[2];
  const float* times  = (const float*)d_in[3];
  const int*   post   = (const int*)d_in[4];
  const float* w_time = (const float*)d_in[5];
  const float* b_time = (const float*)d_in[6];
  const float* Wm1    = (const float*)d_in[7];
  const float* bm1    = (const float*)d_in[8];
  const float* Wm2    = (const float*)d_in[9];
  const float* bm2    = (const float*)d_in[10];
  const float* Wi     = (const float*)d_in[11];
  const float* Wh     = (const float*)d_in[12];
  const float* bi     = (const float*)d_in[13];
  const float* bh     = (const float*)d_in[14];
  const float* Wg     = (const float*)d_in[15];
  const float* We     = (const float*)d_in[16];
  const float* a_src  = (const float*)d_in[17];
  const float* a_dst  = (const float*)d_in[18];
  const float* a_e    = (const float*)d_in[19];
  const float* Wc1    = (const float*)d_in[20];
  const float* bc1    = (const float*)d_in[21];
  const float* Wc2    = (const float*)d_in[22];
  const float* bc2    = (const float*)d_in[23];
  float* out = (float*)d_out;
  const int* ei0 = ei;
  const int* ei1 = ei + E_;

  size_t off = 0;
  auto alloc = [&](size_t bytes) -> void* {
    off = (off + 255) & ~(size_t)255;
    void* p = (char*)d_ws + off;
    off += bytes;
    return p;
  };
  unsigned long long* keys = (unsigned long long*)alloc(8ull * SORT_N);
  int*   order  = (int*)alloc(4ull * E_);
  int*   writer = (int*)alloc(4ull * E_);
  float* te     = (float*)alloc(4ull * E_ * TD);
  float* gte    = (float*)alloc(4ull * TD);
  unsigned* tmaxb = (unsigned*)alloc(4);
  _Float16* memh = (_Float16*)alloc(2ull * N_ * MD);   // TGN node memory, f16
  float* hfeat  = (float*)alloc(4ull * N_ * DD);
  float* outacc = (float*)alloc(4ull * N_ * DD);   // pre1 aliases this (dead before outacc)
  float* pre1   = outacc;
  float* ef     = (float*)alloc(4ull * E_ * DD);
  // f16 weight copies (L2-resident, read 200x by the scan)
  _Float16* Wm1h = (_Float16*)alloc(2ull * 768 * DD);
  _Float16* Wm2h = (_Float16*)alloc(2ull * DD * MD);
  _Float16* Wih  = (_Float16*)alloc(2ull * MD * 768);
  _Float16* Whh  = (_Float16*)alloc(2ull * MD * 768);
  _Float16* Wgh  = (_Float16*)alloc(2ull * 640 * DD);
  _Float16* Weh  = (_Float16*)alloc(2ull * 256 * DD);
  _Float16* Wc1h = (_Float16*)alloc(2ull * DD * 128);
  // scan activations (f16) + GRU gates (f32)
  _Float16* uact = (_Float16*)alloc(2ull * BR * DD);
  _Float16* msgb = (_Float16*)alloc(2ull * BR * MD);
  float* gibuf  = (float*)alloc(4ull * BR * 768);
  float* ghbuf  = (float*)alloc(4ull * BR * 768);
  float* asrcn  = (float*)alloc(4ull * N_);
  float* adstn  = (float*)alloc(4ull * N_);
  float* logit  = (float*)alloc(4ull * E_);
  float* alpha  = (float*)alloc(4ull * E_);
  unsigned* mxenc = (unsigned*)alloc(4ull * N_);
  float* denom  = (float*)alloc(4ull * N_);
  float* q      = (float*)alloc(4ull * P_ * 128);

  auto blks = [](long long n, int b) { return (unsigned)((n + b - 1) / b); };

  // init
  k_fill_f16<<<blks((long long)N_ * MD, 256), 256, 0, stream>>>(memh, 0.f, (long long)N_ * MD);
  k_fill_u32<<<1, 1, 0, stream>>>(tmaxb, 0u, 1);

  // f16 weight conversions
  k_cvt_f16<<<blks(768 * DD, 256), 256, 0, stream>>>(Wm1, Wm1h, 768 * DD);
  k_cvt_f16<<<blks(DD * MD, 256), 256, 0, stream>>>(Wm2, Wm2h, DD * MD);
  k_cvt_f16<<<blks(MD * 768, 256), 256, 0, stream>>>(Wi, Wih, MD * 768);
  k_cvt_f16<<<blks(MD * 768, 256), 256, 0, stream>>>(Wh, Whh, MD * 768);
  k_cvt_f16<<<blks(640 * DD, 256), 256, 0, stream>>>(Wg, Wgh, 640 * DD);
  k_cvt_f16<<<blks(256 * DD, 256), 256, 0, stream>>>(We, Weh, 256 * DD);
  k_cvt_f16<<<blks(DD * 128, 256), 256, 0, stream>>>(Wc1, Wc1h, DD * 128);

  // argsort(edge_times): bitonic on packed (fenc(t), idx) u64 keys
  k_init_keys<<<blks(SORT_N, 256), 256, 0, stream>>>(times, keys);
  for (int k = 2; k <= SORT_N; k <<= 1)
    for (int j = k >> 1; j > 0; j >>= 1)
      k_bitonic<<<blks(SORT_N, 256), 256, 0, stream>>>(keys, j, k);
  k_extract_order<<<blks(E_, 256), 256, 0, stream>>>(keys, order);

  // time encodings
  k_tmax<<<blks(E_, 256), 256, 0, stream>>>(times, tmaxb);
  k_time_enc<<<blks((long long)E_ * TD, 256), 256, 0, stream>>>(times, w_time, b_time, te);
  k_gte<<<1, TD, 0, stream>>>(tmaxb, w_time, b_time, gte);

  // last-write-wins flags per batch
  k_writer<<<blks(E_, 256), 256, 0, stream>>>(order, ei0, writer);

  // mem-independent half of message MLP layer 1 (big WMMA GEMM)
  k_pre1<<<blks((long long)(E_ / 16) * 4 * 32, 256), 256, 0, stream>>>(
      nf, ei0, ea, te, order, Wm1h, bm1, pre1);

  // sequential 200-step memory scan on one WGP (32 waves, async LDS gather)
  k_scan<<<1, 1024, 0, stream>>>(order, ei0, writer, memh, pre1, Wm1h, Wm2h,
                                 bm2, Wih, Whh, bi, bh, uact, msgb, gibuf,
                                 ghbuf);

  // attention embedding (pre1 region now reused as outacc)
  k_fill_f32<<<blks((long long)N_ * DD, 256), 256, 0, stream>>>(outacc, 0.f, (long long)N_ * DD);
  k_fill_f32<<<blks(N_, 256), 256, 0, stream>>>(denom, 0.f, N_);
  k_fill_u32<<<blks(N_, 256), 256, 0, stream>>>(mxenc, ENC_NEG_INF, N_);

  k_hfeat<<<blks((long long)(N_ / 16) * 4 * 32, 256), 256, 0, stream>>>(
      nf, memh, gte, Wgh, hfeat);
  k_ef<<<blks((long long)(E_ / 16) * 4 * 32, 256), 256, 0, stream>>>(ea, te, Weh, ef);
  k_node_dots<<<blks(N_, 256), 256, 0, stream>>>(hfeat, a_src, a_dst, asrcn, adstn);
  k_logit<<<blks(E_, 256), 256, 0, stream>>>(ei0, ei1, ef, a_e, asrcn, adstn, logit, mxenc);
  k_alpha<<<blks(E_, 256), 256, 0, stream>>>(ei1, logit, mxenc, alpha, denom);
  k_scatter<<<blks((long long)E_ * 64, 256), 256, 0, stream>>>(ei0, ei1, alpha, hfeat, ef, outacc);
  k_norm<<<blks((long long)N_ * DD, 256), 256, 0, stream>>>(outacc, denom);

  // classifier head
  k_cls1<<<blks((long long)(P_ / 16) * 2 * 32, 256), 256, 0, stream>>>(
      outacc, post, Wc1h, bc1, q);
  k_cls2<<<blks(P_, 256), 256, 0, stream>>>(q, Wc2, bc2, out);
}
